// GQAWithRopeAttention_49426483642925
// MI455X (gfx1250) — matrix operand order repaired and checked by reading, not compiled, not run on hardware
//
#include <hip/hip_runtime.h>
#include <hip/hip_bf16.h>

// ---------------------------------------------------------------------------
// GQA + RoPE attention (Qwen3-style) for gfx1250 (MI455X), bf16 WMMA pipeline
// with async-to-LDS double-buffered GEMMs.
// ---------------------------------------------------------------------------

typedef __attribute__((ext_vector_type(16))) __bf16 bf16x16;
typedef __attribute__((ext_vector_type(8)))  float  f32x8;

union Frag {
    bf16x16 v;
    uint4   q[2];
};

static __device__ __forceinline__ unsigned short f2bf(float x) {
    unsigned int u = __float_as_uint(x);
    unsigned int r = (u + 0x7FFFu + ((u >> 16) & 1u)) >> 16;  // RNE
    return (unsigned short)r;
}

// gfx1250 async global->LDS copy (ASYNCcnt-tracked), portable via inline asm.
static __device__ __forceinline__ void async_copy_b128(unsigned lds_off,
                                                       const unsigned short* gptr) {
    asm volatile("global_load_async_to_lds_b128 %0, %1, off"
                 :: "v"(lds_off), "v"(gptr) : "memory");
}
static __device__ __forceinline__ void wait_async0() {
    asm volatile("s_wait_asynccnt 0" ::: "memory");
}

#define HD   128
#define NQH  32
#define NKVH 8

// ---------------------------------------------------------------------------
// f32 -> bf16 elementwise convert
// ---------------------------------------------------------------------------
__global__ void __launch_bounds__(256) cvt_f32_bf16(const float* __restrict__ in,
                                                    unsigned short* __restrict__ out,
                                                    size_t n) {
    size_t i = (size_t)blockIdx.x * 256 + threadIdx.x;
    if (i < n) out[i] = f2bf(in[i]);
}

// ---------------------------------------------------------------------------
// bf16 GEMM:  C[M,N] (f32) = A[M,K] * W[N,K]^T
// Block = 8 waves (256 thr). Block tile 256(M) x 64(N) x 32(K).
// A/B staged in LDS via global_load_async_to_lds_b128, double-buffered.
// Each wave computes 32(M) x 64(N): 8 accumulators, 8 WMMAs per K-step.
// ---------------------------------------------------------------------------
#define BM 256
#define BN 64
#define BK 32

__global__ void __launch_bounds__(256) gemm_bf16_wmma(
    const unsigned short* __restrict__ A,   // [M,K] row-major bf16
    const unsigned short* __restrict__ W,   // [N,K] row-major bf16
    float* __restrict__ C,                  // [M,N] f32
    int M, int N, int K)
{
    // stage layout: [A tile BM*BK | B tile BN*BK] per stage, 2 stages
    __shared__ __align__(16) unsigned short lds[2][(BM + BN) * BK];

    const int tid  = threadIdx.x;
    const int lane = tid & 31;
    const int wave = tid >> 5;
    const int mrow = lane & 15;
    const int half = lane >> 4;

    const int m0 = blockIdx.y * BM;
    const int n0 = blockIdx.x * BN;

    f32x8 acc[2][4];
    for (int mi = 0; mi < 2; ++mi)
        for (int t = 0; t < 4; ++t)
            for (int r = 0; r < 8; ++r) acc[mi][t][r] = 0.0f;

    const unsigned short* Ab = A + (size_t)m0 * K;
    const unsigned short* Wb = W + (size_t)n0 * K;

    const unsigned ldsA[2] = { (unsigned)(size_t)&lds[0][0],
                               (unsigned)(size_t)&lds[1][0] };
    const unsigned ldsB[2] = { (unsigned)(size_t)&lds[0][BM * BK],
                               (unsigned)(size_t)&lds[1][BM * BK] };

    // ---- stage loader: 256 threads cooperatively copy 16KB A + 4KB B ----
    // A: 256 rows x 4 segs of 16B; B: 64 rows x 4 segs of 16B.
    const int arow = tid >> 2;          // base row for A tasks (stride 64)
    const int aseg = tid & 3;
    auto stage_load = [&](int s, int k0) {
#pragma unroll
        for (int i = 0; i < 4; ++i) {
            const int row = arow + i * 64;
            async_copy_b128(ldsA[s] + (unsigned)(row * BK + aseg * 8) * 2,
                            Ab + (size_t)row * K + k0 + aseg * 8);
        }
        async_copy_b128(ldsB[s] + (unsigned)(arow * BK + aseg * 8) * 2,
                        Wb + (size_t)(arow)*K + k0 + aseg * 8);
    };

    stage_load(0, 0);
    wait_async0();
    __syncthreads();

    const int nk = K / BK;
    for (int kb = 0; kb < nk; ++kb) {
        const int s = kb & 1;
        if (kb + 1 < nk) stage_load(s ^ 1, (kb + 1) * BK);

        // ---- compute from LDS stage s ----
        const unsigned short* la = &lds[s][0];
        const unsigned short* lb = &lds[s][BM * BK];
        Frag af[2];
#pragma unroll
        for (int mi = 0; mi < 2; ++mi) {
            const unsigned short* ap =
                la + (wave * 32 + mi * 16 + mrow) * BK + half * 8;
            af[mi].q[0] = *(const uint4*)ap;
            af[mi].q[1] = *(const uint4*)(ap + 16);
        }
#pragma unroll
        for (int t = 0; t < 4; ++t) {
            Frag bf;
            const unsigned short* bp = lb + (t * 16 + mrow) * BK + half * 8;
            bf.q[0] = *(const uint4*)bp;
            bf.q[1] = *(const uint4*)(bp + 16);
#pragma unroll
            for (int mi = 0; mi < 2; ++mi)
                acc[mi][t] = __builtin_amdgcn_wmma_f32_16x16x32_bf16(
                    false, af[mi].v, false, bf.v, (short)0, acc[mi][t], false, false);
        }

        wait_async0();
        __syncthreads();
    }

    // ---- store ----
#pragma unroll
    for (int mi = 0; mi < 2; ++mi) {
        const int rbase = m0 + wave * 32 + mi * 16 + half * 8;
#pragma unroll
        for (int t = 0; t < 4; ++t)
#pragma unroll
            for (int r = 0; r < 8; ++r)
                C[(size_t)(rbase + r) * N + n0 + t * 16 + mrow] = acc[mi][t][r];
    }
}

// ---------------------------------------------------------------------------
// Per-head RMSNorm + RoPE, f32 [B*T, H*D] token-major -> bf16 [B,H,T,D]
// One wave per (token, head); lane i handles d = lane + 32*i, i<4.
// ---------------------------------------------------------------------------
__global__ void __launch_bounds__(256) normrope_kernel(
    const float* __restrict__ X,      // [B*T, H*HD]
    const float* __restrict__ w,      // [HD]
    unsigned short* __restrict__ Y,   // [B,H,T,HD]
    int H, int T)
{
    const int gw   = (int)((blockIdx.x * 256 + threadIdx.x) >> 5);
    const int lane = threadIdx.x & 31;
    const int h  = gw % H;
    const int bt = gw / H;
    const int t  = bt % T;
    const int b  = bt / T;

    const float* xp = X + (size_t)bt * (H * HD) + h * HD;
    float x[4];
    float ss = 0.0f;
#pragma unroll
    for (int i = 0; i < 4; ++i) { x[i] = xp[lane + 32 * i]; ss += x[i] * x[i]; }
    ss += __shfl_xor(ss, 1, 32);
    ss += __shfl_xor(ss, 2, 32);
    ss += __shfl_xor(ss, 4, 32);
    ss += __shfl_xor(ss, 8, 32);
    ss += __shfl_xor(ss, 16, 32);
    const float rstd = rsqrtf(ss * (1.0f / 128.0f) + 1e-6f);

    float xn[4];
#pragma unroll
    for (int i = 0; i < 4; ++i) xn[i] = x[i] * rstd * w[lane + 32 * i];

    const float log2theta_div = 19.9315685693f / 64.0f;  // log2(1e6)/64
    float out[4];
#pragma unroll
    for (int i = 0; i < 4; ++i) {
        const int d  = lane + 32 * i;
        const int jj = d & 63;
        const float invf = exp2f(-(float)jj * log2theta_div);
        const float ang  = (float)t * invf;
        const float c = cosf(ang), s = sinf(ang);
        const float rot = (i < 2) ? -xn[i + 2] : xn[i - 2];
        out[i] = xn[i] * c + rot * s;
    }
    unsigned short* yp = Y + (((size_t)(b * H + h)) * T + t) * HD;
#pragma unroll
    for (int i = 0; i < 4; ++i) yp[lane + 32 * i] = f2bf(out[i]);
}

// ---------------------------------------------------------------------------
// V: f32 [B*T, Hkv*D] token-major -> bf16 transposed [B,Hkv,D,T]
// ---------------------------------------------------------------------------
__global__ void __launch_bounds__(256) vtrans_kernel(
    const float* __restrict__ V, unsigned short* __restrict__ Vt,
    int T, size_t n)
{
    size_t i = (size_t)blockIdx.x * 256 + threadIdx.x;
    if (i >= n) return;
    const int col  = (int)(i & (NKVH * HD - 1));
    const size_t tok = i >> 10;                    // / (NKVH*HD) = /1024
    const int t = (int)(tok % T);
    const int b = (int)(tok / T);
    const int h = col >> 7;
    const int d = col & 127;
    Vt[(((size_t)(b * NKVH + h)) * HD + d) * T + t] = f2bf(V[i]);
}

// ---------------------------------------------------------------------------
// Flash attention (causal, GQA). One wave = 16 query rows; block = 8 waves.
// Q bf16 [B,H,T,D], K bf16 [B,Hkv,T,D], Vt bf16 [B,Hkv,D,T],
// O bf16 [B,T,H*D] (token-major, ready for the output projection GEMM).
// ---------------------------------------------------------------------------
__global__ void __launch_bounds__(256) flash_attn_wmma(
    const unsigned short* __restrict__ Q,
    const unsigned short* __restrict__ Kc,
    const unsigned short* __restrict__ Vt,
    unsigned short* __restrict__ O,
    int T)
{
    const int lane = threadIdx.x & 31;
    const int wave = threadIdx.x >> 5;
    const int mrow = lane & 15;
    const int half = lane >> 4;

    const int nqb = T / 128;
    const int bid = blockIdx.x;
    const int b   = bid / (NQH * nqb);
    const int rem = bid % (NQH * nqb);
    const int h   = rem / nqb;
    const int qb  = rem % nqb;
    const int hkv = h >> 2;
    const int q0  = qb * 128 + wave * 16;

    __shared__ __align__(16) unsigned short ldsP[8][16 * 32];
    unsigned short* pbuf = ldsP[wave];

    const unsigned short* qbase = Q  + ((size_t)(b * NQH  + h)   * T + q0) * HD;
    const unsigned short* kbase = Kc + ((size_t)(b * NKVH + hkv) * T) * HD;
    const unsigned short* vbase = Vt + ((size_t)(b * NKVH + hkv) * HD) * T;

    // resident Q fragments (4 chunks of 32 along D)
    Frag qf[4];
#pragma unroll
    for (int c = 0; c < 4; ++c) {
        const unsigned short* qp = qbase + (size_t)mrow * HD + c * 32 + half * 8;
        qf[c].q[0] = *(const uint4*)qp;
        qf[c].q[1] = *(const uint4*)(qp + 16);
    }

    f32x8 o[8];
    for (int dt = 0; dt < 8; ++dt)
        for (int r = 0; r < 8; ++r) o[dt][r] = 0.0f;
    float mrun[8], lrun[8];
    for (int r = 0; r < 8; ++r) { mrun[r] = -1e30f; lrun[r] = 0.0f; }

    const float scale = 0.08838834764831845f;  // 1/sqrt(128)
    const int jmax = (q0 + 15) / 32;

    for (int j = 0; j <= jmax; ++j) {
        const int k0 = j * 32;

        // ---- S = Q * K^T  (two 16x16 tiles over 32 keys) ----
        f32x8 s[2];
        for (int t = 0; t < 2; ++t)
            for (int r = 0; r < 8; ++r) s[t][r] = 0.0f;
#pragma unroll
        for (int t = 0; t < 2; ++t) {
#pragma unroll
            for (int c = 0; c < 4; ++c) {
                Frag kf;
                const unsigned short* kp =
                    kbase + (size_t)(k0 + t * 16 + mrow) * HD + c * 32 + half * 8;
                kf.q[0] = *(const uint4*)kp;
                kf.q[1] = *(const uint4*)(kp + 16);
                s[t] = __builtin_amdgcn_wmma_f32_16x16x32_bf16(
                    false, qf[c].v, false, kf.v, (short)0, s[t], false, false);
            }
        }

        // ---- online softmax in C-fragment layout ----
#pragma unroll
        for (int r = 0; r < 8; ++r) {
            const int qrow = q0 + r + half * 8;
            float s0 = s[0][r] * scale;
            float s1 = s[1][r] * scale;
            if (k0 + mrow      > qrow) s0 = -3e38f;
            if (k0 + 16 + mrow > qrow) s1 = -3e38f;

            float v = fmaxf(s0, s1);
            v = fmaxf(v, __shfl_xor(v, 1, 32));
            v = fmaxf(v, __shfl_xor(v, 2, 32));
            v = fmaxf(v, __shfl_xor(v, 4, 32));
            v = fmaxf(v, __shfl_xor(v, 8, 32));
            const float mnew = fmaxf(mrun[r], v);
            const float corr = __expf(mrun[r] - mnew);
            const float p0 = __expf(s0 - mnew);
            const float p1 = __expf(s1 - mnew);
            float rs = p0 + p1;
            rs += __shfl_xor(rs, 1, 32);
            rs += __shfl_xor(rs, 2, 32);
            rs += __shfl_xor(rs, 4, 32);
            rs += __shfl_xor(rs, 8, 32);
            lrun[r] = lrun[r] * corr + rs;
            mrun[r] = mnew;
#pragma unroll
            for (int dt = 0; dt < 8; ++dt) o[dt][r] *= corr;

            // P -> LDS in [query_row][key] layout (bf16)
            const int qrl = r + 8 * half;
            pbuf[qrl * 32 + mrow]      = f2bf(p0);
            pbuf[qrl * 32 + 16 + mrow] = f2bf(p1);
        }

        // ---- re-load P as A-fragment (same-wave LDS round trip) ----
        Frag pf;
        const unsigned short* pp = pbuf + mrow * 32 + half * 8;
        pf.q[0] = *(const uint4*)pp;
        pf.q[1] = *(const uint4*)(pp + 16);

        // ---- O += P * V  (V pre-transposed: contiguous along keys) ----
#pragma unroll
        for (int dt = 0; dt < 8; ++dt) {
            Frag vf;
            const unsigned short* vp =
                vbase + (size_t)(dt * 16 + mrow) * T + k0 + half * 8;
            vf.q[0] = *(const uint4*)vp;
            vf.q[1] = *(const uint4*)(vp + 16);
            o[dt] = __builtin_amdgcn_wmma_f32_16x16x32_bf16(
                false, pf.v, false, vf.v, (short)0, o[dt], false, false);
        }
    }

    // ---- epilogue: normalize, write bf16 [B,T,H*D] ----
#pragma unroll
    for (int r = 0; r < 8; ++r) {
        const float inv = 1.0f / lrun[r];
        const int t = q0 + r + half * 8;
        unsigned short* orow = O + ((size_t)b * T + t) * (NQH * HD) + h * HD;
#pragma unroll
        for (int dt = 0; dt < 8; ++dt)
            orow[dt * 16 + mrow] = f2bf(o[dt][r] * inv);
    }
}

// ---------------------------------------------------------------------------
// Host-side orchestration
// ---------------------------------------------------------------------------
extern "C" void kernel_launch(void* const* d_in, const int* in_sizes, int n_in,
                              void* d_out, int out_size, void* d_ws, size_t ws_size,
                              hipStream_t stream) {
    const float* hidden = (const float*)d_in[0];  // [B,T,C]
    const float* wq     = (const float*)d_in[1];  // [4096, C]
    const float* wk     = (const float*)d_in[2];  // [1024, C]
    const float* wv     = (const float*)d_in[3];  // [1024, C]
    const float* wo     = (const float*)d_in[4];  // [C, 4096]
    const float* qnw    = (const float*)d_in[5];  // [128]
    const float* knw    = (const float*)d_in[6];  // [128]
    float* out = (float*)d_out;

    const int B = 2, T = 2048, C = 4096;
    const int M = B * T;              // 4096 tokens
    const int NQ = NQH * HD;          // 4096
    const int NK = NKVH * HD;         // 1024

    char* ws = (char*)d_ws;
    size_t off = 0;
    auto alloc = [&](size_t bytes) -> void* {
        void* p = ws + off;
        off += (bytes + 255) & ~(size_t)255;
        return p;
    };

    unsigned short* Xb  = (unsigned short*)alloc((size_t)M * C * 2);
    unsigned short* Wqb = (unsigned short*)alloc((size_t)NQ * C * 2);
    unsigned short* Wkb = (unsigned short*)alloc((size_t)NK * C * 2);
    unsigned short* Wvb = (unsigned short*)alloc((size_t)NK * C * 2);
    unsigned short* Wob = (unsigned short*)alloc((size_t)C * NQ * 2);
    float* Qf = (float*)alloc((size_t)M * NQ * 4);
    float* Kf = (float*)alloc((size_t)M * NK * 4);
    float* Vf = (float*)alloc((size_t)M * NK * 4);
    unsigned short* Qb  = (unsigned short*)alloc((size_t)M * NQ * 2);  // [B,H,T,D]
    unsigned short* Kb  = (unsigned short*)alloc((size_t)M * NK * 2);  // [B,Hkv,T,D]
    unsigned short* Vtb = (unsigned short*)alloc((size_t)M * NK * 2);  // [B,Hkv,D,T]
    unsigned short* Ab  = (unsigned short*)alloc((size_t)M * NQ * 2);  // attn [B,T,H*D]

    auto cvt = [&](const float* src, unsigned short* dst, size_t n) {
        cvt_f32_bf16<<<(unsigned)((n + 255) / 256), 256, 0, stream>>>(src, dst, n);
    };
    cvt(hidden, Xb,  (size_t)M * C);
    cvt(wq,     Wqb, (size_t)NQ * C);
    cvt(wk,     Wkb, (size_t)NK * C);
    cvt(wv,     Wvb, (size_t)NK * C);
    cvt(wo,     Wob, (size_t)C * NQ);

    // QKV projections (async-LDS double-buffered WMMA GEMM)
    gemm_bf16_wmma<<<dim3(NQ / BN, M / BM), 256, 0, stream>>>(Xb, Wqb, Qf, M, NQ, C);
    gemm_bf16_wmma<<<dim3(NK / BN, M / BM), 256, 0, stream>>>(Xb, Wkb, Kf, M, NK, C);
    gemm_bf16_wmma<<<dim3(NK / BN, M / BM), 256, 0, stream>>>(Xb, Wvb, Vf, M, NK, C);

    // RMSNorm + RoPE (Q, K), V transpose-convert
    normrope_kernel<<<(M * NQH) / 8,  256, 0, stream>>>(Qf, qnw, Qb, NQH,  T);
    normrope_kernel<<<(M * NKVH) / 8, 256, 0, stream>>>(Kf, knw, Kb, NKVH, T);
    {
        size_t n = (size_t)M * NK;
        vtrans_kernel<<<(unsigned)((n + 255) / 256), 256, 0, stream>>>(Vf, Vtb, T, n);
    }

    // causal GQA flash attention
    flash_attn_wmma<<<B * NQH * (T / 128), 256, 0, stream>>>(Qb, Kb, Vtb, Ab, T);

    // output projection -> f32 d_out
    gemm_bf16_wmma<<<dim3(C / BN, M / BM), 256, 0, stream>>>(Ab, Wob, out, M, C, NQ);

    (void)in_sizes; (void)n_in; (void)out_size; (void)ws_size;
}
